// Model_49100066128335
// MI455X (gfx1250) — matrix operand order repaired and checked by reading, not compiled, not run on hardware
//
#include <hip/hip_runtime.h>
#include <hip/hip_bf16.h>
#include <math.h>

// ---------------------------------------------------------------------------
// Problem constants (from reference): V,E,HE,HG,T,B = 30000,128,511,512,64,64
// ---------------------------------------------------------------------------
#define TT    64
#define BB    64
#define EE    128
#define HE_S  511          // true encoder hidden
#define HNP   512          // padded hidden (both enc(511->512) and gen(512))
#define G3    (3 * HNP)    // 1536, gate-major stride
#define VOC   30000
#define VOCP  30016        // padded vocab (multiple of 64)
#define NCHUNK (VOCP / 64) // 469 logits chunks

typedef __attribute__((ext_vector_type(16))) _Float16 v16h;
typedef __attribute__((ext_vector_type(8)))  float    v8f;

// ---------------------------------------------------------------------------
// WMMA fragment helpers (wave32, 16x16x32 f16 -> f32).
// A operand: 16x32 tile of a row-major f16 matrix.
//   lane&15 = M row; lane>>4 selects 8-col chunk within each 16-wide K tile;
//   v[0..7] = K tile at k0, v[8..15] = K tile at k0+16.
// B operand: 32x16 tile of a k-major (row = k) f16 matrix, mirrored mapping:
//   lane&15 = N col; lane>>4 selects 8-row chunk within each 16-wide K tile.
// ---------------------------------------------------------------------------
__device__ __forceinline__ v16h load_frag_a(const _Float16* __restrict__ A,
                                            int lda, int m0, int k0, int lane) {
  const int r  = lane & 15;
  const int h8 = (lane >> 4) * 8;
  const _Float16* p = A + (size_t)(m0 + r) * lda + k0 + h8;
  v16h v;
#pragma unroll
  for (int i = 0; i < 8; ++i) { v[i] = p[i]; v[i + 8] = p[16 + i]; }
  return v;
}

__device__ __forceinline__ v16h load_frag_b(const _Float16* __restrict__ B,
                                            int ldb, int k0, int n0, int lane) {
  const int r  = lane & 15;
  const int h8 = (lane >> 4) * 8;
  const _Float16* p = B + (size_t)(k0 + h8) * ldb + n0 + r;
  v16h v;
#pragma unroll
  for (int i = 0; i < 8; ++i) { v[i] = p[(size_t)i * ldb]; v[i + 8] = p[(size_t)(16 + i) * ldb]; }
  return v;
}

__device__ __forceinline__ v8f wmma16(v16h a, v16h b, v8f c) {
  // (neg_a, A, neg_b, B, c_mod, C, reuse_a, reuse_b)
  return __builtin_amdgcn_wmma_f32_16x16x32_f16(false, a, false, b, (short)0, c,
                                                false, false);
}

__device__ __forceinline__ float sigmf(float x) { return 1.f / (1.f + expf(-x)); }

// ---------------------------------------------------------------------------
// One GRU step:  h_next = GRU(x_t, h_prev)   (padded width HNP)
// grid.x = HNP/16 = 32 column tiles; block = 128 (4 waves), wave w -> rows 16w.
// Computes gi = x_t @ Wt_ih (K=128) and gh = h_prev @ Wt_hh (K=HNP) with WMMA,
// then fused gates. Optionally mirrors f32 hidden to h_out (generator states).
// ---------------------------------------------------------------------------
__global__ void __launch_bounds__(128)
gru_step_kernel(const _Float16* __restrict__ x_t,    // BB x EE
                const _Float16* __restrict__ h_prev, // BB x HNP
                const _Float16* __restrict__ Wt_ih,  // EE  x G3 (k-major)
                const _Float16* __restrict__ Wt_hh,  // HNP x G3 (k-major)
                const float*    __restrict__ b_ih,   // G3
                const float*    __restrict__ b_hh,   // G3
                _Float16*       __restrict__ h_next, // BB x HNP
                float*          __restrict__ h_out)  // BB x HNP or nullptr
{
  const int lane = threadIdx.x & 31;
  const int m0   = (threadIdx.x >> 5) * 16;   // wave -> M subtile
  const int n0   = blockIdx.x * 16;           // column subtile

  v8f acc_i[3] = {v8f{}, v8f{}, v8f{}};
  v8f acc_h[3] = {v8f{}, v8f{}, v8f{}};

  // input-side gates: K = 128
  for (int k0 = 0; k0 < EE; k0 += 32) {
    v16h a = load_frag_a(x_t, EE, m0, k0, lane);
#pragma unroll
    for (int g = 0; g < 3; ++g) {
      v16h b = load_frag_b(Wt_ih, G3, k0, g * HNP + n0, lane);
      acc_i[g] = wmma16(a, b, acc_i[g]);
    }
  }
  // hidden-side gates: K = HNP
  for (int k0 = 0; k0 < HNP; k0 += 32) {
    v16h a = load_frag_a(h_prev, HNP, m0, k0, lane);
#pragma unroll
    for (int g = 0; g < 3; ++g) {
      v16h b = load_frag_b(Wt_hh, G3, k0, g * HNP + n0, lane);
      acc_h[g] = wmma16(a, b, acc_h[g]);
    }
  }

  const int col16 = lane & 15;
  const int half  = lane >> 4;
#pragma unroll
  for (int i = 0; i < 8; ++i) {
    const int row = m0 + i + half * 8;        // batch index
    const int col = n0 + col16;               // hidden index
    const float ir = acc_i[0][i] + b_ih[col];
    const float iz = acc_i[1][i] + b_ih[HNP + col];
    const float in_ = acc_i[2][i] + b_ih[2 * HNP + col];
    const float hr = acc_h[0][i] + b_hh[col];
    const float hz = acc_h[1][i] + b_hh[HNP + col];
    const float hn = acc_h[2][i] + b_hh[2 * HNP + col];
    const float r  = sigmf(ir + hr);
    const float z  = sigmf(iz + hz);
    const float nn = tanhf(in_ + r * hn);
    const float hp = (float)h_prev[(size_t)row * HNP + col];
    const float hv = (1.f - z) * nn + z * hp;
    h_next[(size_t)row * HNP + col] = (_Float16)hv;
    if (h_out) h_out[(size_t)row * HNP + col] = hv;
  }
}

// ---------------------------------------------------------------------------
// Logits chunk: block covers 64 vocab columns. 4 waves x 16 cols, each wave
// sweeps 4 M-subtiles with K=512 WMMA. Writes per-chunk (rowmax, sum exp) and
// captures the logit at the label.
// ---------------------------------------------------------------------------
__global__ void __launch_bounds__(128)
logits_kernel(const _Float16* __restrict__ h2,     // BB x HNP
              const _Float16* __restrict__ Wt_out, // HNP x VOCP (k-major)
              const float*    __restrict__ b_out,  // VOCP (pad = -1e4)
              const int*      __restrict__ labels, // BB
              float* __restrict__ cmax,            // NCHUNK x BB
              float* __restrict__ csum,            // NCHUNK x BB
              float* __restrict__ lbl_logit)       // BB
{
  __shared__ float L[BB * 64];
  const int lane  = threadIdx.x & 31;
  const int w     = threadIdx.x >> 5;
  const int nbase = blockIdx.x * 64;
  const int n0    = nbase + w * 16;
  const int col16 = lane & 15;
  const int half  = lane >> 4;

  for (int mt = 0; mt < 4; ++mt) {
    v8f acc = v8f{};
    for (int k0 = 0; k0 < HNP; k0 += 32) {
      v16h a = load_frag_a(h2, HNP, mt * 16, k0, lane);
      v16h b = load_frag_b(Wt_out, VOCP, k0, n0, lane);
      acc = wmma16(a, b, acc);
    }
#pragma unroll
    for (int i = 0; i < 8; ++i) {
      const int row = mt * 16 + i + half * 8;
      const int cl  = w * 16 + col16;
      L[row * 64 + cl] = acc[i] + b_out[nbase + cl];
    }
  }
  __syncthreads();

  if (threadIdx.x < BB) {
    const int b = threadIdx.x;
    float m = -1e30f;
    for (int c = 0; c < 64; ++c) m = fmaxf(m, L[b * 64 + c]);
    float s = 0.f;
    for (int c = 0; c < 64; ++c) s += expf(L[b * 64 + c] - m);
    cmax[blockIdx.x * BB + b] = m;
    csum[blockIdx.x * BB + b] = s;
    const int lb = labels[b];
    if (lb >= nbase && lb < nbase + 64) lbl_logit[b] = L[b * 64 + (lb - nbase)];
  }
}

// Merge chunk stats -> per-step masked loss (one block, BB threads)
__global__ void loss_step_kernel(const float* __restrict__ cmax,
                                 const float* __restrict__ csum,
                                 const float* __restrict__ lbl_logit,
                                 const int*   __restrict__ labels,
                                 float* __restrict__ loss_t) {
  __shared__ float sn[BB], sm[BB];
  const int b = threadIdx.x;
  float M = -1e30f;
  for (int c = 0; c < NCHUNK; ++c) M = fmaxf(M, cmax[c * BB + b]);
  float S = 0.f;
  for (int c = 0; c < NCHUNK; ++c) S += csum[c * BB + b] * expf(cmax[c * BB + b] - M);
  const float lse  = M + logf(S);
  const float nll  = lse - lbl_logit[b];
  const float mask = (labels[b] != 0) ? 1.f : 0.f;
  sn[b] = nll * mask;
  sm[b] = mask;
  __syncthreads();
  for (int st = BB / 2; st > 0; st >>= 1) {
    if (b < st) { sn[b] += sn[b + st]; sm[b] += sm[b + st]; }
    __syncthreads();
  }
  if (b == 0) *loss_t = sn[0] / fmaxf(sm[0], 1.f);
}

__global__ void avg_loss_kernel(const float* __restrict__ losses, float* __restrict__ out) {
  __shared__ float s[TT];
  const int t = threadIdx.x;
  s[t] = losses[t];
  __syncthreads();
  for (int st = TT / 2; st > 0; st >>= 1) {
    if (t < st) s[t] += s[t + st];
    __syncthreads();
  }
  if (t == 0) *out = s[0] / (float)TT;
}

// ---------------------------------------------------------------------------
// Prep kernels: f32 -> f16 conversions with transpose to k-major + padding.
// ---------------------------------------------------------------------------
// dst[k][g*Np + j] = (j<Hsrc && k<Ksrc) ? src[(g*Hsrc+j)*Ksrc + k] : 0
__global__ void prep_gate_weight(const float* __restrict__ src, _Float16* __restrict__ dst,
                                 int Ksrc, int Kp, int Hsrc, int Np) {
  const int i = blockIdx.x * blockDim.x + threadIdx.x;
  const int total = Kp * 3 * Np;
  if (i >= total) return;
  const int k = i / (3 * Np);
  const int c = i - k * 3 * Np;
  const int g = c / Np;
  const int j = c - g * Np;
  const float v = (j < Hsrc && k < Ksrc) ? src[(size_t)(g * Hsrc + j) * Ksrc + k] : 0.f;
  dst[i] = (_Float16)v;
}

__global__ void prep_wout(const float* __restrict__ W, _Float16* __restrict__ dst) {
  const int i = blockIdx.x * blockDim.x + threadIdx.x;
  if (i >= HNP * VOCP) return;
  const int k = i / VOCP;
  const int v = i - k * VOCP;
  const float val = (v < VOC) ? W[(size_t)v * HNP + k] : 0.f;
  dst[i] = (_Float16)val;
}

__global__ void prep_bias_pad(const float* __restrict__ src, float* __restrict__ dst,
                              int Hsrc, int Np) {
  const int i = blockIdx.x * blockDim.x + threadIdx.x;
  if (i >= 3 * Np) return;
  const int g = i / Np, j = i - g * Np;
  dst[i] = (j < Hsrc) ? src[g * Hsrc + j] : 0.f;
}

__global__ void prep_bout(const float* __restrict__ b, float* __restrict__ dst) {
  const int i = blockIdx.x * blockDim.x + threadIdx.x;
  if (i >= VOCP) return;
  dst[i] = (i < VOC) ? b[i] : -1.0e4f;
}

// Embedding gather for encoder inputs and shifted generator inputs
__global__ void embed_kernel(const int* __restrict__ tok, const float* __restrict__ emb,
                             _Float16* __restrict__ xe, _Float16* __restrict__ xg) {
  const int i = blockIdx.x * blockDim.x + threadIdx.x;
  if (i >= TT * BB * EE) return;
  const int tb = i >> 7;          // token slot (t*BB + b)
  const int k  = i & (EE - 1);
  const int t  = tb / BB;
  xe[i] = (_Float16)emb[(size_t)tok[tb] * EE + k];
  const int tg = (t == 0) ? 1 /*GO*/ : tok[tb - BB];
  xg[i] = (_Float16)emb[(size_t)tg * EE + k];
}

__global__ void zero_h_kernel(_Float16* __restrict__ h) {
  const int i = blockIdx.x * blockDim.x + threadIdx.x;
  if (i < BB * HNP) h[i] = (_Float16)0.f;
}

// ---------------------------------------------------------------------------
// Host launcher
// ---------------------------------------------------------------------------
extern "C" void kernel_launch(void* const* d_in, const int* in_sizes, int n_in,
                              void* d_out, int out_size, void* d_ws, size_t ws_size,
                              hipStream_t stream) {
  (void)in_sizes; (void)n_in; (void)out_size; (void)ws_size;

  const int*   input    = (const int*)  d_in[0];   // T x B
  const float* emb      = (const float*)d_in[1];   // V x E
  const float* W_ih_e   = (const float*)d_in[2];   // 1533 x 128
  const float* W_hh_e   = (const float*)d_in[3];   // 1533 x 511
  const float* b_ih_e   = (const float*)d_in[4];   // 1533
  const float* b_hh_e   = (const float*)d_in[5];   // 1533
  const float* W_ih_g   = (const float*)d_in[6];   // 1536 x 128
  const float* W_hh_g   = (const float*)d_in[7];   // 1536 x 512
  const float* b_ih_g   = (const float*)d_in[8];   // 1536
  const float* b_hh_g   = (const float*)d_in[9];   // 1536
  const float* W_out    = (const float*)d_in[10];  // 30000 x 512
  const float* b_out    = (const float*)d_in[11];  // 30000
  float* out = (float*)d_out;                      // T*B*HG hidden states + 1 loss

  // workspace bump allocator
  char* wp = (char*)d_ws;
  auto alloc = [&](size_t bytes) -> void* {
    void* p = (void*)wp;
    wp += (bytes + 255) & ~(size_t)255;
    return p;
  };
  _Float16* Wt_ih_e = (_Float16*)alloc((size_t)EE  * G3 * 2);
  _Float16* Wt_hh_e = (_Float16*)alloc((size_t)HNP * G3 * 2);
  _Float16* Wt_ih_g = (_Float16*)alloc((size_t)EE  * G3 * 2);
  _Float16* Wt_hh_g = (_Float16*)alloc((size_t)HNP * G3 * 2);
  _Float16* Wt_out  = (_Float16*)alloc((size_t)HNP * VOCP * 2);
  float*    bi_e    = (float*)alloc(G3 * 4);
  float*    bh_e    = (float*)alloc(G3 * 4);
  float*    b_out_p = (float*)alloc(VOCP * 4);
  _Float16* xh_enc  = (_Float16*)alloc((size_t)TT * BB * EE * 2);
  _Float16* xh_gen  = (_Float16*)alloc((size_t)TT * BB * EE * 2);
  _Float16* hA      = (_Float16*)alloc((size_t)BB * HNP * 2);
  _Float16* hB      = (_Float16*)alloc((size_t)BB * HNP * 2);
  float*    cmax    = (float*)alloc((size_t)NCHUNK * BB * 4);
  float*    csum    = (float*)alloc((size_t)NCHUNK * BB * 4);
  float*    lbl     = (float*)alloc(BB * 4);
  float*    losses  = (float*)alloc(TT * 4);

  const int TB = 256;
  auto nb = [](int total, int tb) { return (total + tb - 1) / tb; };

  // ---- prep ----
  prep_gate_weight<<<nb(EE  * G3, TB), TB, 0, stream>>>(W_ih_e, Wt_ih_e, EE,  EE,  HE_S, HNP);
  prep_gate_weight<<<nb(HNP * G3, TB), TB, 0, stream>>>(W_hh_e, Wt_hh_e, HE_S, HNP, HE_S, HNP);
  prep_gate_weight<<<nb(EE  * G3, TB), TB, 0, stream>>>(W_ih_g, Wt_ih_g, EE,  EE,  HNP, HNP);
  prep_gate_weight<<<nb(HNP * G3, TB), TB, 0, stream>>>(W_hh_g, Wt_hh_g, HNP, HNP, HNP, HNP);
  prep_wout<<<nb(HNP * VOCP, TB), TB, 0, stream>>>(W_out, Wt_out);
  prep_bias_pad<<<nb(G3, TB), TB, 0, stream>>>(b_ih_e, bi_e, HE_S, HNP);
  prep_bias_pad<<<nb(G3, TB), TB, 0, stream>>>(b_hh_e, bh_e, HE_S, HNP);
  prep_bout<<<nb(VOCP, TB), TB, 0, stream>>>(b_out, b_out_p);
  embed_kernel<<<nb(TT * BB * EE, TB), TB, 0, stream>>>(input, emb, xh_enc, xh_gen);
  zero_h_kernel<<<nb(BB * HNP, TB), TB, 0, stream>>>(hA);

  // ---- encoder recurrence (h padded to 512; pad lane provably stays 0) ----
  _Float16* hp = hA;
  _Float16* hn = hB;
  for (int t = 0; t < TT; ++t) {
    gru_step_kernel<<<HNP / 16, 128, 0, stream>>>(
        xh_enc + (size_t)t * BB * EE, hp, Wt_ih_e, Wt_hh_e, bi_e, bh_e, hn, nullptr);
    _Float16* tmp = hp; hp = hn; hn = tmp;
  }
  // hp == latent = concat(h_enc, 0)  (padded column is exactly the zero concat)

  // ---- generator recurrence + fused streamed log-softmax loss ----
  for (int t = 0; t < TT; ++t) {
    gru_step_kernel<<<HNP / 16, 128, 0, stream>>>(
        xh_gen + (size_t)t * BB * EE, hp, Wt_ih_g, Wt_hh_g, b_ih_g, b_hh_g, hn,
        out + (size_t)t * BB * HNP);
    logits_kernel<<<NCHUNK, 128, 0, stream>>>(hn, Wt_out, b_out_p, input + t * BB,
                                              cmax, csum, lbl);
    loss_step_kernel<<<1, BB, 0, stream>>>(cmax, csum, lbl, input + t * BB, losses + t);
    _Float16* tmp = hp; hp = hn; hn = tmp;
  }

  avg_loss_kernel<<<1, TT, 0, stream>>>(losses, out + (size_t)TT * BB * HNP);
}
